// CoreAttention_58050777973525
// MI455X (gfx1250) — matrix operand order repaired and checked by reading, not compiled, not run on hardware
//
#include <hip/hip_runtime.h>
#include <stdint.h>

// ---------------------------------------------------------------------------
// CoreAttention for MI455X (gfx1250, wave32, WMMA + Tensor Data Mover).
//   out = softmax( (q*sqrt(D)) @ k^T + bias, masked->0 ) @ v
// B=2, S=2048, H=16, D=64. fp32 in/out, f16 WMMA internally (fp32 accum).
// Bias (537 MB stream, the roofline term) + mask are DMA'd global->LDS by the
// TDM, double-buffered under compute; K/V^T staged manually (small traffic).
// ---------------------------------------------------------------------------

typedef _Float16 v16h __attribute__((ext_vector_type(16)));
typedef float    v8f  __attribute__((ext_vector_type(8)));
typedef unsigned int v4u __attribute__((ext_vector_type(4)));
typedef int          v8i __attribute__((ext_vector_type(8)));
typedef int          v4i __attribute__((ext_vector_type(4)));

union FragH { uint32_t u[8]; v16h v; };

__device__ __forceinline__ uint32_t pack2h(float a, float b) {
  union { _Float16 h[2]; uint32_t u; } t;
  t.h[0] = (_Float16)a; t.h[1] = (_Float16)b;
  return t.u;
}

__device__ __forceinline__ v8f wmma16x16x32(const FragH& a, const FragH& b, v8f c) {
  return __builtin_amdgcn_wmma_f32_16x16x32_f16(false, a.v, false, b.v,
                                                (short)0, c, false, false);
}

// Raw LDS byte offset of a __shared__ object (generic -> addrspace(3)).
__device__ __forceinline__ uint32_t lds_off_of(const void* p) {
  return (uint32_t)(__SIZE_TYPE__)(const __attribute__((address_space(3))) char*)p;
}

// Issue a 2D TDM tile load: global (row-major, stride0 elements between rows)
// -> LDS, with D# pad fields adding 1 dword after every tile row.
//   dsize_code: 0=1B, 2=4B.  padint_code: row bytes = (1<<padint_code)*8.
__device__ __forceinline__ void tdm_load_2d(uint32_t lds_off, const void* gptr,
                                            uint32_t tile0, uint32_t tile1,
                                            uint32_t stride0, uint32_t dsize_code,
                                            uint32_t padint_code) {
  const uint64_t ga = (uint64_t)(uintptr_t)gptr;
  v4u g0;
  g0[0] = 1u;                                        // count=1 (valid), user mode
  g0[1] = lds_off;                                   // lds_addr
  g0[2] = (uint32_t)ga;                              // global_addr[31:0]
  g0[3] = (uint32_t)((ga >> 32) & 0x01FFFFFFu) | 0x80000000u; // addr[56:32] | type=2
  v8i g1;
  g1[0] = (int)((dsize_code << 16) | (1u << 20) | (padint_code << 22)); // pad_amount=0 -> 1 dword
  g1[1] = (int)(0x8000u << 16);                      // tensor_dim0 = 0x8000 (low16)
  g1[2] = (int)(0x8000u << 16);                      // tdim0 hi=0 | tensor_dim1=0x8000 (low16)
  g1[3] = (int)(tile0 << 16);                        // tdim1 hi=0 | tile_dim0
  g1[4] = (int)tile1;                                // tile_dim1 | tile_dim2=0
  g1[5] = (int)stride0;                              // tensor_dim0_stride[31:0] (elements)
  g1[6] = 0;                                         // stride0 hi | dim1_stride lo
  g1[7] = 0;
  const v4i gz4 = {0, 0, 0, 0};
  const v8i gz8 = {0, 0, 0, 0, 0, 0, 0, 0};
  __builtin_amdgcn_tensor_load_to_lds(g0, g1, gz4, gz4, gz8, 0);
}

constexpr int Bn = 2, Sn = 2048, Hn = 16, Dn = 64;
constexpr int WAVES = 8;             // 256 threads = 8 wave32
constexpr int QTILE = 16;            // queries per wave
constexpr int QBLK  = WAVES * QTILE; // 128 queries per block
constexpr int KSTEP = 32;            // keys per main-loop step

__global__ __launch_bounds__(256)
void fa_fwd_kernel(const float* __restrict__ Q, const float* __restrict__ K,
                   const float* __restrict__ V, const float* __restrict__ Bias,
                   const unsigned char* __restrict__ Mask,
                   float* __restrict__ Out)
{
  // K tile, f16 packed pairs over D: [32 keys][32 dwords + 1 pad]
  __shared__ uint32_t s_k[32 * 33];
  // V^T, f16 packed pairs over keys: [64 d][16 dwords + 1 pad]
  __shared__ uint32_t s_vt[64 * 17];
  // TDM-filled, double-buffered, per-wave tiles (row-major [q][key], padded):
  __shared__ float         s_bias2[2][WAVES][16 * 33];  // 33-dword row pitch
  __shared__ unsigned char s_mask2[2][WAVES][16 * 36];  // 36-byte row pitch

  _Float16* vtH = (_Float16*)s_vt;

  const int tid  = threadIdx.x;
  const int wave = tid >> 5;
  const int lane = tid & 31;
  const int hk   = lane >> 4;
  const int ln   = lane & 15;

  const int nqb = Sn / QBLK;                 // 16 query blocks per (b,h)
  const int bid = blockIdx.x;
  const int b   = bid / (Hn * nqb);
  const int rem = bid % (Hn * nqb);
  const int h   = rem / nqb;
  const int qb  = rem % nqb;
  const int q0w = qb * QBLK + wave * QTILE;

  const size_t HD = (size_t)Hn * Dn;
  const float* Krow0 = K + (((size_t)b * Sn) * Hn + h) * Dn;
  const float* Vrow0 = V + (((size_t)b * Sn) * Hn + h) * Dn;
  const float* BiasB = Bias + (((size_t)b * Hn + h) * Sn) * Sn;
  const unsigned char* MaskB = Mask + ((size_t)b * Sn) * Sn;

  // ---- Q^T as B-matrix fragments (loaded once, scaled by sqrt(D)=8) ----
  FragH bql, bqh;
  {
    const float2* q2 = (const float2*)(Q + (((size_t)b * Sn + q0w + ln) * Hn + h) * Dn);
#pragma unroll
    for (int vv = 0; vv < 8; vv++) {
      float2 lo = q2[hk * 8 + vv];
      float2 hi = q2[hk * 8 + vv + 16];
      bql.u[vv] = pack2h(8.0f * lo.x, 8.0f * lo.y);
      bqh.u[vv] = pack2h(8.0f * hi.x, 8.0f * hi.y);
    }
  }

  float mrun = -INFINITY, lrun = 0.0f;
  v8f acc[4];
#pragma unroll
  for (int t = 0; t < 4; t++)
#pragma unroll
    for (int r = 0; r < 8; r++) acc[t][r] = 0.0f;

  // ---- prologue: TDM the kb=0 bias+mask tiles into buffer 0 ----
  tdm_load_2d(lds_off_of(&s_bias2[0][wave][0]), BiasB + (size_t)q0w * Sn,
              KSTEP, QTILE, Sn, 2, 4);
  tdm_load_2d(lds_off_of(&s_mask2[0][wave][0]), MaskB + (size_t)q0w * Sn,
              KSTEP, QTILE, Sn, 0, 2);

  for (int it = 0, kb = 0; kb < Sn; kb += KSTEP, ++it) {
    const int buf = it & 1;

    // ================= stage K / V^T (block-cooperative, sync) ==========
#pragma unroll
    for (int i = 0; i < 4; i++) {
      const int row = wave * 4 + i;
      const float2* kr = (const float2*)(Krow0 + (size_t)(kb + row) * HD);
      float2 kk = kr[lane];
      s_k[row * 33 + lane] = pack2h(kk.x, kk.y);
      const float* vr = Vrow0 + (size_t)(kb + row) * HD;
      float v0 = vr[lane];
      float v1 = vr[lane + 32];
      vtH[(size_t)lane * 34 + row]        = (_Float16)v0;
      vtH[(size_t)(lane + 32) * 34 + row] = (_Float16)v1;
    }
    __syncthreads();

    // ====== issue next tile's TDM pair, then wait for current pair ======
    if (kb + KSTEP < Sn) {
      const int nb = buf ^ 1;
      const int nkb = kb + KSTEP;
      tdm_load_2d(lds_off_of(&s_bias2[nb][wave][0]),
                  BiasB + (size_t)q0w * Sn + nkb, KSTEP, QTILE, Sn, 2, 4);
      tdm_load_2d(lds_off_of(&s_mask2[nb][wave][0]),
                  MaskB + (size_t)q0w * Sn + nkb, KSTEP, QTILE, Sn, 0, 2);
      __builtin_amdgcn_s_wait_tensorcnt(2);  // in-order: current pair complete
    } else {
      __builtin_amdgcn_s_wait_tensorcnt(0);
    }

    // ================= S^T = K_tile @ Q^T  (two 16-key tiles) ============
    v8f c0, c1;
    {
      FragH al, ah;
#pragma unroll
      for (int vv = 0; vv < 8; vv++) {
        const int col = hk * 8 + vv;
        al.u[vv] = s_k[ln * 33 + col];
        ah.u[vv] = s_k[ln * 33 + col + 16];
      }
      v8f z = {0.f, 0.f, 0.f, 0.f, 0.f, 0.f, 0.f, 0.f};
      c0 = wmma16x16x32(al, bql, z);
      c0 = wmma16x16x32(ah, bqh, c0);
#pragma unroll
      for (int vv = 0; vv < 8; vv++) {
        const int col = hk * 8 + vv;
        al.u[vv] = s_k[(16 + ln) * 33 + col];
        ah.u[vv] = s_k[(16 + ln) * 33 + col + 16];
      }
      c1 = wmma16x16x32(al, bql, z);
      c1 = wmma16x16x32(ah, bqh, c1);
    }

    // ====== scores + bias (+mask->0) and online softmax (per lane q) =====
    const float* bwp = &s_bias2[buf][wave][0];
    const unsigned char* mwp = &s_mask2[buf][wave][0];
    float sv[2][8];
    float mloc = -INFINITY;
#pragma unroll
    for (int t = 0; t < 2; t++)
#pragma unroll
      for (int r = 0; r < 8; r++) {
        const int key = t * 16 + r + 8 * hk;
        const float bv = bwp[ln * 33 + key];
        const unsigned char mk = mwp[ln * 36 + key];
        const float qk = (t == 0) ? c0[r] : c1[r];
        const float s  = mk ? 0.0f : (qk + bv);
        sv[t][r] = s;
        mloc = fmaxf(mloc, s);
      }
    mloc = fmaxf(mloc, __shfl_xor(mloc, 16));
    const float mnew  = fmaxf(mrun, mloc);
    const float alpha = __expf(mrun - mnew);
    lrun *= alpha;
#pragma unroll
    for (int t = 0; t < 4; t++) acc[t] *= alpha;

    float ps = 0.0f;
#pragma unroll
    for (int t = 0; t < 2; t++)
#pragma unroll
      for (int r = 0; r < 8; r++) {
        const float p = __expf(sv[t][r] - mnew);
        sv[t][r] = p;
        ps += p;
      }
    ps  += __shfl_xor(ps, 16);
    lrun += ps;
    mrun  = mnew;

    // ====== build P^T as B-fragment (cross-half exchange + f16 pack) =====
    float ex0[8], ex1[8];
#pragma unroll
    for (int r = 0; r < 8; r++) {
      ex0[r] = __shfl_xor(sv[0][r], 16);
      ex1[r] = __shfl_xor(sv[1][r], 16);
    }
    FragH bp;
#pragma unroll
    for (int vv = 0; vv < 4; vv++) {
      const float a = hk ? ex1[2 * vv]     : sv[0][2 * vv];
      const float c = hk ? ex1[2 * vv + 1] : sv[0][2 * vv + 1];
      bp.u[vv] = pack2h(a, c);
    }
#pragma unroll
    for (int vv = 4; vv < 8; vv++) {
      const int j = 2 * (vv - 4);
      const float a = hk ? sv[1][j]     : ex0[j];
      const float c = hk ? sv[1][j + 1] : ex0[j + 1];
      bp.u[vv] = pack2h(a, c);
    }

    // ================= out^T += V^T_tile @ P^T  (4 d-tiles) ==============
#pragma unroll
    for (int t = 0; t < 4; t++) {
      FragH av;
      const int d = t * 16 + ln;
#pragma unroll
      for (int vv = 0; vv < 8; vv++) {
        const int col = hk * 8 + vv;
        av.u[vv] = s_vt[d * 17 + col];
      }
      acc[t] = wmma16x16x32(av, bp, acc[t]);
    }
    __syncthreads();
  }

  // ==================== normalize and store out[b, q, h, d] ==============
  const float il = 1.0f / lrun;
#pragma unroll
  for (int t = 0; t < 4; t++) {
    const size_t off = (((size_t)b * Sn + q0w + ln) * Hn + h) * Dn + t * 16 + hk * 8;
    float4 o0 = {acc[t][0] * il, acc[t][1] * il, acc[t][2] * il, acc[t][3] * il};
    float4 o1 = {acc[t][4] * il, acc[t][5] * il, acc[t][6] * il, acc[t][7] * il};
    *(float4*)(Out + off)     = o0;
    *(float4*)(Out + off + 4) = o1;
  }
}

extern "C" void kernel_launch(void* const* d_in, const int* in_sizes, int n_in,
                              void* d_out, int out_size, void* d_ws, size_t ws_size,
                              hipStream_t stream) {
  const float* q    = (const float*)d_in[0];
  const float* k    = (const float*)d_in[1];
  const float* v    = (const float*)d_in[2];
  const float* bias = (const float*)d_in[3];
  const unsigned char* mask = (const unsigned char*)d_in[4];
  float* out = (float*)d_out;

  const int blocks = Bn * Hn * (Sn / QBLK);  // 512
  fa_fwd_kernel<<<dim3(blocks), dim3(256), 0, stream>>>(q, k, v, bias, mask, out);
}
